// Fused_FSRCNN_net_48704929136843
// MI455X (gfx1250) — matrix operand order repaired and checked by reading, not compiled, not run on hardware
//
#include <hip/hip_runtime.h>

// ============================================================================
// FSRCNN on gfx1250 via implicit-GEMM WMMA (v_wmma_f32_16x16x32_f16).
//
// Activations fp16 NHWC, channels padded to 16 (S=12->16, 3->16) or 64
// (D=56->64). Every conv is a GEMM: M = 16-pixel tile, N = 16-out-channel
// tile, K = taps*Cpad chunked by 32 (fully unrolled, compile-time taps).
// Transposed tail conv is sub-pixel decomposed into a 5x5 conv with N = 12
// (3 ch x 4 phases), pixel-shuffled into the f32 output.
// Boundary handling: clamped-address loads + per-lane select (no exec
// branches) so the scheduler can pipeline loads across WMMAs.
// ============================================================================

typedef __attribute__((ext_vector_type(16))) _Float16 v16h;
typedef __attribute__((ext_vector_type(8)))  _Float16 v8h;
typedef __attribute__((ext_vector_type(8)))  float    v8f;

#define DEVI __device__ __forceinline__

// ---------------------------------------------------------------------------
// Input packing: NCHW f32 (3 ch) -> [img][pixel][16] f16, channels 3..15 = 0
// ---------------------------------------------------------------------------
__global__ void pack_input_kernel(const float* __restrict__ x,
                                  _Float16* __restrict__ a0) {
  int t = blockIdx.x * blockDim.x + threadIdx.x;      // 0 .. 1024*1024-1
  int img = t >> 10;
  int p   = t & 1023;
  const float* xi = x + (size_t)img * 3072;           // 3*32*32
  _Float16* o = a0 + (size_t)t * 16;
#pragma unroll
  for (int c = 0; c < 16; ++c)
    o[c] = (c < 3) ? (_Float16)xi[c * 1024 + p] : (_Float16)0.f;
}

// ---------------------------------------------------------------------------
// Weight packing into per-lane-contiguous B fragments (ISA 7.12.2):
//   lanes 0-15 : col n = lane,    halfs j=0..15 -> B[k0+{0..7,16..23}][n]
//   lanes 16-31: col n = lane-16, halfs j=0..15 -> B[k0+{8..15,24..31}][n]
// B[k][n]: k = tap*Cpad + ci, weight = w[n][ci][ky][kx] (OIHW), pad -> 0.
// dst element (nt, chunk, lane, j) at ((nt*nChunks+chunk)*32+lane)*16 + j.
// ---------------------------------------------------------------------------
__global__ void pack_w_kernel(const float* __restrict__ src,
                              _Float16* __restrict__ dst,
                              int Cout, int Cin, int KH, int KW,
                              int clog2, int nChunks, int nTiles) {
  int t = blockIdx.x * blockDim.x + threadIdx.x;
  int total = nTiles * nChunks * 32;
  if (t >= total) return;
  int lane  = t & 31;
  int chunk = (t >> 5) % nChunks;
  int nt    = t / (nChunks * 32);
  int hi    = (lane >> 4) & 1;
  int n     = nt * 16 + (lane & 15);
  int cmask = (1 << clog2) - 1;
  _Float16* d = dst + (size_t)t * 16;
#pragma unroll 1
  for (int j = 0; j < 16; ++j) {
    int k   = chunk * 32 + hi * 8 + j + (j >= 8 ? 8 : 0);
    int tap = k >> clog2;
    int ci  = k & cmask;
    float v = 0.f;
    if (n < Cout && ci < Cin && tap < KH * KW)
      v = src[(((size_t)n * Cin + ci) * KH + tap / KW) * KW + (tap % KW)];
    d[j] = (_Float16)v;
  }
}

// Tail (ConvTranspose k=9,s=2,p=4,op=1) sub-pixel weights:
// phase (ry,rx), out ch co -> n=(ry*2+rx)*3+co; tap (dy,dx) in [-2..2]^2;
// ky = ry==0 ? 4-2*dy : 5-2*dy (same for kx); w_tail is (56,3,9,9) IOHW.
__global__ void pack_w_tail_kernel(const float* __restrict__ tw,
                                   _Float16* __restrict__ dst) {
  int t = blockIdx.x * blockDim.x + threadIdx.x;      // 50 chunks * 32 lanes
  if (t >= 50 * 32) return;
  int lane  = t & 31;
  int chunk = t >> 5;
  int hi    = (lane >> 4) & 1;
  int n     = lane & 15;
  _Float16* d = dst + (size_t)t * 16;
#pragma unroll 1
  for (int j = 0; j < 16; ++j) {
    int k   = chunk * 32 + hi * 8 + j + (j >= 8 ? 8 : 0);
    int tap = k >> 6;
    int ci  = k & 63;
    float v = 0.f;
    if (n < 12 && ci < 56 && tap < 25) {
      int dy = tap / 5 - 2, dx = tap % 5 - 2;
      int ry = n / 6, rx = (n / 3) & 1, co = n % 3;
      int ky = (ry == 0) ? (4 - 2 * dy) : (5 - 2 * dy);
      int kx = (rx == 0) ? (4 - 2 * dx) : (5 - 2 * dx);
      if ((unsigned)ky < 9u && (unsigned)kx < 9u)
        v = tw[(((size_t)ci * 3 + co) * 9 + ky) * 9 + kx];
    }
    d[j] = (_Float16)v;
  }
}

// ---------------------------------------------------------------------------
// A-fragment gather (16x32, ISA 7.12.2):
//   lanes 0-15 : row M = lane,    halfs = K {k0+0..7,  k0+16..23}
//   lanes 16-31: row M = lane-16, halfs = K {k0+8..15, k0+24..31}
// Each 8-half group lives inside one tap's padded-channel run. `chunk` is a
// compile-time constant after unrolling, so tap/dy/dx fold away (kbase is a
// multiple of 16 and hi*8 < 16, so hi never perturbs tap for Cpad 16/64).
// Out-of-image taps: load from clamped offset, select zero (no exec branch).
// ---------------------------------------------------------------------------
template<int CLOG2, int KW, int PAD, int NTAPS>
DEVI v16h load_a_frag(const _Float16* __restrict__ in_img,
                      int p0, int lane, int chunk) {
  int m  = lane & 15;
  int p  = p0 + m;
  int y  = p >> 5, x = p & 31;
  int hi = (lane >> 4) & 1;
  v16h a;
#pragma unroll
  for (int g = 0; g < 2; ++g) {
    const int kbase  = chunk * 32 + g * 16;            // constant
    const int tap    = kbase >> CLOG2;                 // constant
    const int cibase = kbase & ((1 << CLOG2) - 1);     // constant
    int ci = cibase + hi * 8;
    v8h v = {};
    if (tap < NTAPS) {                                 // constant-folded
      const int dy = tap / KW - PAD;                   // constant
      const int dx = tap % KW - PAD;                   // constant
      if (dy == 0 && dx == 0) {
        v = *(const v8h*)(in_img + (((size_t)((y << 5) + x)) << CLOG2) + ci);
      } else {
        int yy = y + dy, xx = x + dx;
        bool ok = ((unsigned)yy < 32u) & ((unsigned)xx < 32u);
        int off = (yy << 5) + xx;
        off = off < 0 ? 0 : (off > 1023 ? 1023 : off); // safe clamped addr
        v8h t0 = *(const v8h*)(in_img + (((size_t)off) << CLOG2) + ci);
        v = ok ? t0 : v;                               // v_cndmask, no branch
      }
    }
#pragma unroll
    for (int i = 0; i < 8; ++i) a[g * 8 + i] = v[i];
  }
  return a;
}

// ---------------------------------------------------------------------------
// Generic conv layer: one wave = 16 pixels x (NT*16) out channels.
// grid = 1024 imgs * 16 blocks, block = 128 threads (4 wave32).
// Fully unrolled K loop; dual accumulators when NT==1 to keep the matrix
// pipe fed. Epilogue fuses bias + optional PReLU, stores fp16 NHWC (padded
// lanes get 0 since their packed weights and bias are 0).
// ---------------------------------------------------------------------------
template<int CLOG2, int KW, int PAD, int NTAPS, int NC, int NT, int COLOG2,
         bool PRELU>
__global__ __launch_bounds__(128)
void conv_wmma_kernel(const _Float16* __restrict__ in,
                      _Float16* __restrict__ out,
                      const _Float16* __restrict__ wpack,
                      const float* __restrict__ bias,
                      const float* __restrict__ alpha,
                      int Cout) {
  int img   = blockIdx.x >> 4;
  int mtile = ((blockIdx.x & 15) << 2) + (threadIdx.x >> 5);
  int lane  = threadIdx.x & 31;
  int p0    = mtile << 4;
  const _Float16* ii = in + ((size_t)img << (10 + CLOG2));

  __builtin_prefetch(wpack + (size_t)lane * 16, 0, 1);  // global_prefetch

  constexpr int NACC = (NT == 1) ? 2 : 1;
  v8f acc[NT * NACC] = {};
#pragma unroll
  for (int c = 0; c < NC; ++c) {
    v16h a = load_a_frag<CLOG2, KW, PAD, NTAPS>(ii, p0, lane, c);
#pragma unroll
    for (int t = 0; t < NT; ++t) {
      v16h b = *(const v16h*)(wpack + (((size_t)t * NC + c) * 32 + lane) * 16);
      int ai = t * NACC + (c & (NACC - 1));
      acc[ai] = __builtin_amdgcn_wmma_f32_16x16x32_f16(
          false, a, false, b, (short)0, acc[ai], false, false);
    }
  }

  int hi = lane >> 4, n0 = lane & 15;
  _Float16* oimg = out + ((size_t)img << (10 + COLOG2));
#pragma unroll
  for (int t = 0; t < NT; ++t) {
    v8f r = acc[t * NACC];
    if constexpr (NACC == 2) r += acc[t * NACC + 1];
    int n    = t * 16 + n0;
    float bv = (n < Cout) ? bias[n] : 0.f;
    float av = 0.f;
    if constexpr (PRELU) av = (n < Cout) ? alpha[n] : 0.f;
#pragma unroll
    for (int q = 0; q < 8; ++q) {
      float v = r[q] + bv;                      // D-frag: M = hi*8+q, N = n
      if constexpr (PRELU) { if (v < 0.f) v *= av; }
      int p = p0 + hi * 8 + q;
      oimg[((size_t)p << COLOG2) + n] = (_Float16)v;
    }
  }
}

// ---------------------------------------------------------------------------
// Tail: 5x5 conv over 64-padded channels, N = 12 (3 ch x 4 phases, pad 16),
// K = 25*64 = 1600 -> 50 chunks, fully unrolled, dual accumulators.
// Epilogue pixel-shuffles into 3x64x64 f32 output (+ bias).
// ---------------------------------------------------------------------------
__global__ __launch_bounds__(128)
void tail_wmma_kernel(const _Float16* __restrict__ in,
                      float* __restrict__ out,
                      const _Float16* __restrict__ wpack,
                      const float* __restrict__ bias) {
  int img   = blockIdx.x >> 4;
  int mtile = ((blockIdx.x & 15) << 2) + (threadIdx.x >> 5);
  int lane  = threadIdx.x & 31;
  int p0    = mtile << 4;
  const _Float16* ii = in + ((size_t)img << 16);   // 1024 px * 64 ch

  __builtin_prefetch(wpack + (size_t)lane * 16, 0, 1);

  v8f acc0 = {}, acc1 = {};
#pragma unroll
  for (int c = 0; c < 50; ++c) {
    v16h a = load_a_frag<6, 5, 2, 25>(ii, p0, lane, c);
    v16h b = *(const v16h*)(wpack + (((size_t)c) * 32 + lane) * 16);
    if (c & 1)
      acc1 = __builtin_amdgcn_wmma_f32_16x16x32_f16(
          false, a, false, b, (short)0, acc1, false, false);
    else
      acc0 = __builtin_amdgcn_wmma_f32_16x16x32_f16(
          false, a, false, b, (short)0, acc0, false, false);
  }
  v8f acc = acc0 + acc1;

  int hi = lane >> 4, n = lane & 15;
  if (n < 12) {
    int ry = n / 6, rx = (n / 3) & 1, co = n % 3;
    float bv = bias[co];
    float* ob = out + ((size_t)img * 3 + co) * 4096;
#pragma unroll
    for (int q = 0; q < 8; ++q) {
      int p  = p0 + hi * 8 + q;
      int py = p >> 5, px = p & 31;
      ob[(2 * py + ry) * 64 + (2 * px + rx)] = acc[q] + bv;
    }
  }
}

// ---------------------------------------------------------------------------
// Launcher
// ---------------------------------------------------------------------------
extern "C" void kernel_launch(void* const* d_in, const int* in_sizes, int n_in,
                              void* d_out, int out_size, void* d_ws, size_t ws_size,
                              hipStream_t stream) {
  const float* x      = (const float*)d_in[0];
  const float* head_w = (const float*)d_in[1];
  const float* head_b = (const float*)d_in[2];
  const float* head_a = (const float*)d_in[3];
  const float* b0_w   = (const float*)d_in[4];
  const float* b0_b   = (const float*)d_in[5];
  const float* b0_a   = (const float*)d_in[6];
  const float* b1_w   = (const float*)d_in[7];
  const float* b1_b   = (const float*)d_in[8];
  const float* b2_w   = (const float*)d_in[9];
  const float* b2_b   = (const float*)d_in[10];
  const float* b3_w   = (const float*)d_in[11];
  const float* b3_b   = (const float*)d_in[12];
  const float* b4_w   = (const float*)d_in[13];
  const float* b4_b   = (const float*)d_in[14];
  const float* b5_a   = (const float*)d_in[15];
  const float* b6_w   = (const float*)d_in[16];
  const float* b6_b   = (const float*)d_in[17];
  const float* b6_a   = (const float*)d_in[18];
  const float* tail_w = (const float*)d_in[19];
  const float* tail_b = (const float*)d_in[20];

  char* ws = (char*)d_ws;
  const size_t NPIX = (size_t)1024 * 1024;   // imgs * pixels

  // Weight-pack region (first 1 MB), then activations.
  _Float16* wp_head = (_Float16*)ws;               // 4*13*32*16 = 26624 halfs
  _Float16* wp_b0   = wp_head + 26624;             // 1*2*32*16  = 1024
  _Float16* wp_b1   = wp_b0   + 1024;              // 1*5*32*16  = 2560
  _Float16* wp_b2   = wp_b1   + 2560;
  _Float16* wp_b3   = wp_b2   + 2560;
  _Float16* wp_b4   = wp_b3   + 2560;
  _Float16* wp_b6   = wp_b4   + 2560;              // 4*1*32*16  = 2048
  _Float16* wp_tail = wp_b6   + 2048;              // 1*50*32*16 = 25600

  _Float16* A0  = (_Float16*)(ws + (1 << 20));     // [img][px][16]  32 MB
  _Float16* H64 = A0  + NPIX * 16;                 // [img][px][64] 128 MB
  _Float16* HA  = H64 + NPIX * 64;                 // [img][px][16]  32 MB
  _Float16* HB  = HA  + NPIX * 16;                 // [img][px][16]  32 MB

  // ---- pack inputs & weights -------------------------------------------
  pack_input_kernel<<<4096, 256, 0, stream>>>(x, A0);
  pack_w_kernel<<<7, 256, 0, stream>>>(head_w, wp_head, 56,  3, 5, 5, 4, 13, 4);
  pack_w_kernel<<<1, 256, 0, stream>>>(b0_w,   wp_b0,   12, 56, 1, 1, 6,  2, 1);
  pack_w_kernel<<<1, 256, 0, stream>>>(b1_w,   wp_b1,   12, 12, 3, 3, 4,  5, 1);
  pack_w_kernel<<<1, 256, 0, stream>>>(b2_w,   wp_b2,   12, 12, 3, 3, 4,  5, 1);
  pack_w_kernel<<<1, 256, 0, stream>>>(b3_w,   wp_b3,   12, 12, 3, 3, 4,  5, 1);
  pack_w_kernel<<<1, 256, 0, stream>>>(b4_w,   wp_b4,   12, 12, 3, 3, 4,  5, 1);
  pack_w_kernel<<<1, 256, 0, stream>>>(b6_w,   wp_b6,   56, 12, 1, 1, 4,  1, 1);
  pack_w_tail_kernel<<<7, 256, 0, stream>>>(tail_w, wp_tail);

  // ---- network ---------------------------------------------------------
  const dim3 grid(16384), block(128);
  // head: 5x5, 16->64ch, K=25*16=400 (13 chunks), PReLU
  conv_wmma_kernel<4, 5, 2, 25, 13, 4, 6, true><<<grid, block, 0, stream>>>(
      A0, H64, wp_head, head_b, head_a, 56);
  // b0: 1x1, 64->16ch, K=64 (2 chunks), PReLU
  conv_wmma_kernel<6, 1, 0, 1, 2, 1, 4, true><<<grid, block, 0, stream>>>(
      H64, HA, wp_b0, b0_b, b0_a, 12);
  // b1..b4: 3x3, 16->16ch, K=9*16=144 (5 chunks)
  conv_wmma_kernel<4, 3, 1, 9, 5, 1, 4, false><<<grid, block, 0, stream>>>(
      HA, HB, wp_b1, b1_b, nullptr, 12);
  conv_wmma_kernel<4, 3, 1, 9, 5, 1, 4, false><<<grid, block, 0, stream>>>(
      HB, HA, wp_b2, b2_b, nullptr, 12);
  conv_wmma_kernel<4, 3, 1, 9, 5, 1, 4, false><<<grid, block, 0, stream>>>(
      HA, HB, wp_b3, b3_b, nullptr, 12);
  conv_wmma_kernel<4, 3, 1, 9, 5, 1, 4, true><<<grid, block, 0, stream>>>(
      HB, HA, wp_b4, b4_b, b5_a, 12);
  // b6: 1x1, 16->64ch, K=16 (1 chunk, upper half zero), PReLU
  conv_wmma_kernel<4, 1, 0, 1, 1, 4, 6, true><<<grid, block, 0, stream>>>(
      HA, H64, wp_b6, b6_b, b6_a, 56);
  // tail: sub-pixel 5x5, 64ch -> 12 phase-channels, shuffle to f32 out
  tail_wmma_kernel<<<grid, block, 0, stream>>>(
      H64, (float*)d_out, wp_tail, tail_b);
}
